// GraphAttentionLayer_16741782520216
// MI455X (gfx1250) — matrix operand order repaired and checked by reading, not compiled, not run on hardware
//
#include <hip/hip_runtime.h>

#define BATCH 8
#define NVERT 2048
#define FIN   128
#define FOUT  64
#define ALPHA 0.2f
#define NEGINF -9.0e15f

typedef __attribute__((ext_vector_type(16))) _Float16 v16h;
typedef __attribute__((ext_vector_type(8)))  float    v8f;

union AFrag { v16h v; _Float16 h[16]; };
union CFrag { v8f  v; float f[8]; };
union H8    { float4 f4; _Float16 h[8]; };
union V16f  { float4 q[4]; float f[16]; };
union V16i  { int4   q[4]; int   v[16]; };

// ---------------------------------------------------------------------------
// Kernel 1: Wh = h @ W  (f16 WMMA, f32 accum).  One wave = 16 rows x 64 cols.
// Also writes a transposed f16 copy WhT[b][col][n] for fast B-fragment loads.
// ---------------------------------------------------------------------------
__global__ void k_wh(const float* __restrict__ h, const float* __restrict__ W,
                     float* __restrict__ Wh, _Float16* __restrict__ WhT) {
  const int lane = threadIdx.x & 31;
  const int tile = blockIdx.x * 4 + (threadIdx.x >> 5); // 16-row tile over B*N
  const int m16  = lane & 15;
  const int hlf  = lane >> 4;
  const int g    = 8 * hlf;
  const int row0 = tile * 16;
  const int arow = row0 + m16;

  CFrag acc[4];
  #pragma unroll
  for (int nt = 0; nt < 4; ++nt)
    #pragma unroll
    for (int q = 0; q < 8; ++q) acc[nt].f[q] = 0.0f;

  for (int kb = 0; kb < FIN; kb += 32) {
    // A fragment: h[arow][kb+g .. +7] and [kb+g+16 .. +23], f32 -> f16
    const float* hp = h + (size_t)arow * FIN + kb + g;
    V16f ht;
    ht.q[0] = *(const float4*)(hp + 0);
    ht.q[1] = *(const float4*)(hp + 4);
    ht.q[2] = *(const float4*)(hp + 16);
    ht.q[3] = *(const float4*)(hp + 20);
    AFrag a;
    #pragma unroll
    for (int q = 0; q < 16; ++q) a.h[q] = (_Float16)ht.f[q];

    #pragma unroll
    for (int nt = 0; nt < 4; ++nt) {
      // B fragment: W[kb + 8*hlf + (i&7) + 16*(i>>3)][nt*16 + m16], i = 0..15
      AFrag bf;
      #pragma unroll
      for (int i = 0; i < 16; ++i)
        bf.h[i] = (_Float16)W[(size_t)(kb + 8 * hlf + (i & 7) + ((i >> 3) << 4)) * FOUT + nt * 16 + m16];
      acc[nt].v = __builtin_amdgcn_wmma_f32_16x16x32_f16(
          false, a.v, false, bf.v, (short)0, acc[nt].v, false, false);
    }
  }

  #pragma unroll
  for (int nt = 0; nt < 4; ++nt)
    #pragma unroll
    for (int rr = 0; rr < 8; ++rr) {
      const int r   = row0 + rr + 8 * hlf;   // global flattened row (b*N+n)
      const int col = nt * 16 + m16;
      const float v = acc[nt].f[rr];
      Wh[(size_t)r * FOUT + col] = v;
      const int b = r >> 11;                 // / 2048
      const int n = r & 2047;
      WhT[((size_t)b * FOUT + col) * NVERT + n] = (_Float16)v;
    }
}

// ---------------------------------------------------------------------------
// Kernel 2: f1 = Wh @ a[:64], f2 = Wh @ a[64:]   (one thread per row)
// ---------------------------------------------------------------------------
__global__ void k_f1f2(const float* __restrict__ Wh, const float* __restrict__ a,
                       float* __restrict__ f1, float* __restrict__ f2) {
  const int r = blockIdx.x * blockDim.x + threadIdx.x;
  if (r >= BATCH * NVERT) return;
  const float* w = Wh + (size_t)r * FOUT;
  float s1 = 0.0f, s2 = 0.0f;
  #pragma unroll 8
  for (int c = 0; c < FOUT; ++c) {
    const float v = w[c];
    s1 += v * a[c];
    s2 += v * a[FOUT + c];
  }
  f1[r] = s1;
  f2[r] = s2;
}

// ---------------------------------------------------------------------------
// Kernel 3: per-row online softmax stats (max, sumexp).  One wave per row.
// ---------------------------------------------------------------------------
__global__ void k_stats(const int* __restrict__ adj, const float* __restrict__ f1,
                        const float* __restrict__ f2,
                        float* __restrict__ rowmax, float* __restrict__ rowsum) {
  const int wave = (int)((blockIdx.x * blockDim.x + threadIdx.x) >> 5);
  const int lane = threadIdx.x & 31;
  if (wave >= BATCH * NVERT) return;
  const int b = wave >> 11;
  const int i = wave & 2047;
  const float f1i = f1[wave];
  const int*   ar = adj + (size_t)i * NVERT;
  const float* fb = f2 + (size_t)b * NVERT;

  float m = -3.0e38f, l = 0.0f;
  for (int j = lane; j < NVERT; j += 32) {
    float x = f1i + fb[j];
    x = (x > 0.0f) ? x : ALPHA * x;
    const float e = (ar[j] > 0) ? x : NEGINF;
    const float mn = fmaxf(m, e);
    l = l * __expf(m - mn) + __expf(e - mn);
    m = mn;
  }
  #pragma unroll
  for (int off = 16; off > 0; off >>= 1) {
    const float m2 = __shfl_xor(m, off, 32);
    const float l2 = __shfl_xor(l, off, 32);
    const float mn = fmaxf(m, m2);
    l = l * __expf(m - mn) + l2 * __expf(m2 - mn);
    m = mn;
  }
  if (lane == 0) { rowmax[wave] = m; rowsum[wave] = l; }
}

// ---------------------------------------------------------------------------
// Kernel 4: fused attention matmul.  One wave = 16 output rows x 64 cols.
// P regenerated on the fly in WMMA A layout; B fragments from WhT (2x b128).
// out = elu(P @ Wh) + bias
// ---------------------------------------------------------------------------
__global__ void k_attn(const int* __restrict__ adj, const float* __restrict__ mask,
                       const float* __restrict__ f1, const float* __restrict__ f2,
                       const float* __restrict__ rowmax, const float* __restrict__ rowsum,
                       const _Float16* __restrict__ WhT, const float* __restrict__ bias,
                       float* __restrict__ out) {
  const int lane = threadIdx.x & 31;
  const int tile = blockIdx.x * 4 + (threadIdx.x >> 5);
  const int m16  = lane & 15;
  const int hlf  = lane >> 4;
  const int g    = 8 * hlf;
  const int row0 = tile * 16;
  const int b    = row0 >> 11;
  const int r    = row0 + m16;       // this lane's A-matrix row (b*N+n)
  const int i    = r & 2047;

  const float f1r  = f1[r];
  const float mrow = rowmax[r];
  const float linv = 1.0f / rowsum[r];

  const float* fb = f2   + (size_t)b * NVERT;
  const int*   ar = adj  + (size_t)i * NVERT;
  const float* mk = mask + (size_t)i * NVERT;

  CFrag acc[4];
  #pragma unroll
  for (int nt = 0; nt < 4; ++nt)
    #pragma unroll
    for (int q = 0; q < 8; ++q) acc[nt].f[q] = 0.0f;

  for (int j0 = 0; j0 < NVERT; j0 += 32) {
    if (j0 + 32 < NVERT) {
      __builtin_prefetch(ar + j0 + 32 + g, 0, 3);   // global_prefetch_b8
      __builtin_prefetch(mk + j0 + 32 + g, 0, 3);
    }
    // Gather the 16 attention logits this lane owns (A layout: K = g+0..7, g+16..23)
    V16f f2t, mkt; V16i adt;
    f2t.q[0] = *(const float4*)(fb + j0 + g);      f2t.q[1] = *(const float4*)(fb + j0 + g + 4);
    f2t.q[2] = *(const float4*)(fb + j0 + g + 16); f2t.q[3] = *(const float4*)(fb + j0 + g + 20);
    adt.q[0] = *(const int4*)(ar + j0 + g);        adt.q[1] = *(const int4*)(ar + j0 + g + 4);
    adt.q[2] = *(const int4*)(ar + j0 + g + 16);   adt.q[3] = *(const int4*)(ar + j0 + g + 20);
    mkt.q[0] = *(const float4*)(mk + j0 + g);      mkt.q[1] = *(const float4*)(mk + j0 + g + 4);
    mkt.q[2] = *(const float4*)(mk + j0 + g + 16); mkt.q[3] = *(const float4*)(mk + j0 + g + 20);

    AFrag a;
    #pragma unroll
    for (int q = 0; q < 16; ++q) {
      float x = f1r + f2t.f[q];
      x = (x > 0.0f) ? x : ALPHA * x;
      const float e = (adt.v[q] > 0) ? x : NEGINF;
      const float p = __expf(e - mrow) * mkt.f[q] * linv;
      a.h[q] = (_Float16)p;
    }

    #pragma unroll
    for (int nt = 0; nt < 4; ++nt) {
      // B fragment: WhT[b][nt*16+m16][j0 + 8*hlf + 0..7, +16..23] -> 2 b128
      const _Float16* wp = WhT + ((size_t)b * FOUT + nt * 16 + m16) * NVERT + j0 + 8 * hlf;
      H8 lo, hi;
      lo.f4 = *(const float4*)(wp);
      hi.f4 = *(const float4*)(wp + 16);
      AFrag bf;
      #pragma unroll
      for (int q = 0; q < 8; ++q) { bf.h[q] = lo.h[q]; bf.h[8 + q] = hi.h[q]; }
      acc[nt].v = __builtin_amdgcn_wmma_f32_16x16x32_f16(
          false, a.v, false, bf.v, (short)0, acc[nt].v, false, false);
    }
  }

  #pragma unroll
  for (int nt = 0; nt < 4; ++nt)
    #pragma unroll
    for (int rr = 0; rr < 8; ++rr) {
      const int orow = row0 + rr + 8 * hlf;
      const int col  = nt * 16 + m16;
      float x = acc[nt].f[rr];
      x = (x > 0.0f) ? x : (__expf(x) - 1.0f);   // elu
      out[(size_t)orow * FOUT + col] = x + bias[col];
    }
}

// ---------------------------------------------------------------------------
extern "C" void kernel_launch(void* const* d_in, const int* in_sizes, int n_in,
                              void* d_out, int out_size, void* d_ws, size_t ws_size,
                              hipStream_t stream) {
  const float* h    = (const float*)d_in[0];
  const int*   adj  = (const int*)  d_in[1];
  const float* W    = (const float*)d_in[2];
  const float* a    = (const float*)d_in[3];
  const float* mask = (const float*)d_in[4];
  const float* bias = (const float*)d_in[5];
  float* out = (float*)d_out;

  const size_t ROWS = (size_t)BATCH * NVERT;          // 16384
  char* ws = (char*)d_ws;
  size_t off = 0;
  float* Wh     = (float*)(ws + off); off += ROWS * FOUT * sizeof(float);   // 4 MB
  float* f1     = (float*)(ws + off); off += ROWS * sizeof(float);
  float* f2     = (float*)(ws + off); off += ROWS * sizeof(float);
  float* rowmax = (float*)(ws + off); off += ROWS * sizeof(float);
  float* rowsum = (float*)(ws + off); off += ROWS * sizeof(float);
  _Float16* WhT = (_Float16*)(ws + off);              // 2 MB

  const int tiles = (int)(ROWS / 16);                 // 1024

  k_wh   <<<tiles / 4,        128, 0, stream>>>(h, W, Wh, WhT);
  k_f1f2 <<<(int)(ROWS / 256), 256, 0, stream>>>(Wh, a, f1, f2);
  k_stats<<<(int)(ROWS / 4),  128, 0, stream>>>(adj, f1, f2, rowmax, rowsum);
  k_attn <<<tiles / 4,        128, 0, stream>>>(adj, mask, f1, f2, rowmax, rowsum,
                                                WhT, bias, out);
}
